// AttentionCacheAdapter_23553600651571
// MI455X (gfx1250) — compile-verified, matrix operand order checked by reading, not run on hardware
//
#include <hip/hip_runtime.h>
#include <hip/hip_bf16.h>
#include <stdint.h>

// ---------------------------------------------------------------------------
// Types for CDNA5 WMMA (wave32): A/B = v16h (16 halves = 8 VGPRs), C/D = v8f.
// ---------------------------------------------------------------------------
typedef _Float16 v16h __attribute__((ext_vector_type(16)));
typedef _Float16 h8v  __attribute__((ext_vector_type(8)));
typedef float    v8f  __attribute__((ext_vector_type(8)));
typedef int      v4i  __attribute__((ext_vector_type(4)));

union F16x16 { v16h v; h8v h[2]; };

__device__ __forceinline__ v8f wmma_f16(v16h a, v16h b, v8f c) {
  // D = A(16x32 f16) x B(32x16 f16) + C(16x16 f32)
  return __builtin_amdgcn_wmma_f32_16x16x32_f16(
      /*neg_a=*/false, a, /*neg_b=*/false, b,
      /*c_mod=*/(short)0, c, /*reuse_a=*/false, /*reuse_b=*/false);
}

__device__ __forceinline__ v8f v8f_zero() {
  v8f z = {0.f,0.f,0.f,0.f,0.f,0.f,0.f,0.f};
  return z;
}

// Optional CDNA5 async global->LDS staging (guarded; falls back to sync path).
#if defined(__has_builtin)
#if __has_builtin(__builtin_amdgcn_global_load_async_to_lds_b128) && \
    __has_builtin(__builtin_amdgcn_s_wait_asynccnt)
#define USE_ASYNC_LDS 1
#endif
#endif
#ifndef USE_ASYNC_LDS
#define USE_ASYNC_LDS 0
#endif

#define AS1 __attribute__((address_space(1)))
#define AS3 __attribute__((address_space(3)))

#if USE_ASYNC_LDS
// Copy 16 bytes global -> LDS via GLOBAL_LOAD_ASYNC_TO_LDS_B128 (ASYNCcnt).
// Builtin signature (from hipcc diagnostic): (int4 AS1*, int4 AS3*, imm, imm).
__device__ __forceinline__ void async_copy_b128(const void* gsrc, void* ldst) {
  AS1 v4i* g = (AS1 v4i*)(AS1 void*)(void*)gsrc;
  AS3 v4i* l = (AS3 v4i*)(AS3 void*)ldst;
  __builtin_amdgcn_global_load_async_to_lds_b128(g, l, 0, 0);
}
#endif

// Problem constants
#define S_TOK 2048
#define D_MODEL 4096
#define HQ 32
#define KVH 8
#define DH 128
#define HD (HQ*DH)     // 4096
#define KVD (KVH*DH)   // 1024

// ---------------------------------------------------------------------------
// Kernel 0: f32 -> f16 convert (hidden_states)
// ---------------------------------------------------------------------------
__global__ __launch_bounds__(256) void cvt_f32_f16(const float* __restrict__ src,
                                                   _Float16* __restrict__ dst) {
  size_t i = ((size_t)blockIdx.x * blockDim.x + threadIdx.x) * 8;
  float4 a = *(const float4*)(src + i);
  float4 b = *(const float4*)(src + i + 4);
  h8v h;
  h[0]=(_Float16)a.x; h[1]=(_Float16)a.y; h[2]=(_Float16)a.z; h[3]=(_Float16)a.w;
  h[4]=(_Float16)b.x; h[5]=(_Float16)b.y; h[6]=(_Float16)b.z; h[7]=(_Float16)b.w;
  *(h8v*)(dst + i) = h;
}

// ---------------------------------------------------------------------------
// WMMA GEMM:  C[M,N] = A[M,K](f16, row-major) @ B[N,K](f32, row-major)^T
// Block tile 128x128, BK=32, 256 threads = 8 waves in 2(M)x4(N) grid,
// wave tile 64x32 (8 WMMA per K-step per wave, 12 ds_load_b128).
// LDS stride 40 halves (80B) keeps ds_load_b128 fragment reads 16B aligned.
// A tile is staged via GLOBAL_LOAD_ASYNC_TO_LDS when the toolchain has it.
// ---------------------------------------------------------------------------
#define GLDS 40

template <bool OUTF32>
__global__ __launch_bounds__(256) void gemm_wmma(const _Float16* __restrict__ A,
                                                 const float* __restrict__ B,
                                                 _Float16* __restrict__ C16,
                                                 float* __restrict__ C32,
                                                 int K, int lda, int ldb, int ldc) {
  __shared__ _Float16 As[128 * GLDS];  // 10 KB
  __shared__ _Float16 Bs[128 * GLDS];  // 10 KB

  const int bm = blockIdx.y * 128;
  const int bn = blockIdx.x * 128;
  const int tid = threadIdx.x;
  const int lane = tid & 31;
  const int wave = tid >> 5;
  const int wm = (wave >> 2) * 64;  // 2 waves along M, 64 rows each
  const int wn = (wave & 3) * 32;   // 4 waves along N, 32 cols each

  const int col = lane & 15;
  const int off = (lane >> 4) * 8;  // 16-bit A/B fragment K-offset per half-wave

  v8f acc[4][2];
#pragma unroll
  for (int i = 0; i < 4; ++i)
#pragma unroll
    for (int j = 0; j < 2; ++j) acc[i][j] = v8f_zero();

  // A staging map: thread -> (row = tid>>1, 16 halves at (tid&1)*16)
  const int arow = tid >> 1, acg = (tid & 1) * 16;
  // B staging map: thread -> (row = tid>>1, 16 floats at (tid&1)*16)
  const int brow = tid >> 1, bc0 = (tid & 1) * 16;

  for (int k0 = 0; k0 < K; k0 += 32) {
    // ---- global -> LDS (A, f16 direct copy) ----
    {
      const _Float16* gsrc = A + (size_t)(bm + arow) * lda + k0 + acg;
#if USE_ASYNC_LDS
      async_copy_b128(gsrc,     &As[arow * GLDS + acg]);
      async_copy_b128(gsrc + 8, &As[arow * GLDS + acg + 8]);
#else
      const h8v* src = (const h8v*)gsrc;
      *(h8v*)&As[arow * GLDS + acg]     = src[0];
      *(h8v*)&As[arow * GLDS + acg + 8] = src[1];
#endif
    }
    // ---- global -> LDS (B, f32 -> f16 convert through VGPRs) ----
    {
      const float* gb = B + (size_t)(bn + brow) * ldb + k0 + bc0;
      const float4* src = (const float4*)gb;
      float4 f0 = src[0], f1 = src[1], f2 = src[2], f3 = src[3];
      if (k0 + 32 < K) __builtin_prefetch(gb + 32, 0, 0);  // next B K-tile
      h8v h0, h1;
      h0[0]=(_Float16)f0.x; h0[1]=(_Float16)f0.y; h0[2]=(_Float16)f0.z; h0[3]=(_Float16)f0.w;
      h0[4]=(_Float16)f1.x; h0[5]=(_Float16)f1.y; h0[6]=(_Float16)f1.z; h0[7]=(_Float16)f1.w;
      h1[0]=(_Float16)f2.x; h1[1]=(_Float16)f2.y; h1[2]=(_Float16)f2.z; h1[3]=(_Float16)f2.w;
      h1[4]=(_Float16)f3.x; h1[5]=(_Float16)f3.y; h1[6]=(_Float16)f3.z; h1[7]=(_Float16)f3.w;
      *(h8v*)&Bs[brow * GLDS + bc0]     = h0;
      *(h8v*)&Bs[brow * GLDS + bc0 + 8] = h1;
    }
#if USE_ASYNC_LDS
    __builtin_amdgcn_s_wait_asynccnt(0);
#endif
    __syncthreads();

    // ---- fragments (ISA 16-bit layout: halves[0..7]=K off.., halves[8..15]=K 16+off..) ----
    F16x16 af[4], bf[2];
#pragma unroll
    for (int i = 0; i < 4; ++i) {
      int r = wm + i * 16 + col;
      af[i].h[0] = *(const h8v*)&As[r * GLDS + off];
      af[i].h[1] = *(const h8v*)&As[r * GLDS + 16 + off];
    }
#pragma unroll
    for (int j = 0; j < 2; ++j) {
      int r = wn + j * 16 + col;
      bf[j].h[0] = *(const h8v*)&Bs[r * GLDS + off];
      bf[j].h[1] = *(const h8v*)&Bs[r * GLDS + 16 + off];
    }
#pragma unroll
    for (int i = 0; i < 4; ++i)
#pragma unroll
      for (int j = 0; j < 2; ++j)
        acc[i][j] = wmma_f16(af[i].v, bf[j].v, acc[i][j]);
    __syncthreads();
  }

  // ---- store: C element (vgpr r, lane) = [m0 + r + (lane>>4)*8][n0 + (lane&15)] ----
  const int hi = lane >> 4;
#pragma unroll
  for (int i = 0; i < 4; ++i)
#pragma unroll
    for (int j = 0; j < 2; ++j) {
      int m0 = bm + wm + i * 16 + hi * 8;
      int n0 = bn + wn + j * 16 + col;
#pragma unroll
      for (int r = 0; r < 8; ++r) {
        float val = acc[i][j][r];
        if (OUTF32) C32[(size_t)(m0 + r) * ldc + n0] = val;
        else        C16[(size_t)(m0 + r) * ldc + n0] = (_Float16)val;
      }
    }
}

// ---------------------------------------------------------------------------
// RoPE on Q (in place, f16 buffer [S, H*DH]); one thread per rotation pair.
// ---------------------------------------------------------------------------
__global__ __launch_bounds__(256) void rope_q_kernel(_Float16* __restrict__ Q) {
  size_t idx = (size_t)blockIdx.x * blockDim.x + threadIdx.x;  // S*H*64
  int i = (int)(idx & 63);
  int h = (int)((idx >> 6) & 31);
  int s = (int)(idx >> 11);
  _Float16* p = Q + (size_t)s * HD + h * DH;
  float x1 = (float)p[i], x2 = (float)p[i + 64];
  float invf = __powf(10000.f, -(float)i * (1.f / 64.f));
  float fr = (float)s * invf;
  float c = __cosf(fr), sn = __sinf(fr);
  p[i]      = (_Float16)(x1 * c - x2 * sn);
  p[i + 64] = (_Float16)(x2 * c + x1 * sn);
}

// ---------------------------------------------------------------------------
// RoPE + int8 quant-dequant on K (in place, [S, KVH*DH]); one wave per (s,kv).
// ---------------------------------------------------------------------------
__global__ __launch_bounds__(256) void rope_quant_k_kernel(_Float16* __restrict__ Kb) {
  int wid  = (int)(((size_t)blockIdx.x * blockDim.x + threadIdx.x) >> 5);  // 0..S*KVH-1
  int lane = threadIdx.x & 31;
  int kv = wid & (KVH - 1);
  int s  = wid >> 3;
  _Float16* p = Kb + (size_t)s * KVD + kv * DH;

  float x0 = (float)p[lane],      x1 = (float)p[lane + 64];
  float x2 = (float)p[lane + 32], x3 = (float)p[lane + 96];

  float f0 = (float)s * __powf(10000.f, -(float)lane        * (1.f / 64.f));
  float f1 = (float)s * __powf(10000.f, -(float)(lane + 32) * (1.f / 64.f));
  float c0 = __cosf(f0), s0 = __sinf(f0);
  float c1 = __cosf(f1), s1 = __sinf(f1);

  float y[4];
  y[0] = x0 * c0 - x1 * s0;  // dim lane
  y[1] = x1 * c0 + x0 * s0;  // dim lane+64
  y[2] = x2 * c1 - x3 * s1;  // dim lane+32
  y[3] = x3 * c1 + x2 * s1;  // dim lane+96

  float amax = fmaxf(fmaxf(fabsf(y[0]), fabsf(y[1])), fmaxf(fabsf(y[2]), fabsf(y[3])));
#pragma unroll
  for (int m = 16; m >= 1; m >>= 1) amax = fmaxf(amax, __shfl_xor(amax, m, 32));
  float sc = fmaxf(amax * (1.f / 127.f), 1e-8f);
  float rs = 1.f / sc;
#pragma unroll
  for (int t = 0; t < 4; ++t)
    y[t] = fminf(fmaxf(rintf(y[t] * rs), -127.f), 127.f) * sc;

  p[lane]      = (_Float16)y[0];
  p[lane + 64] = (_Float16)y[1];
  p[lane + 32] = (_Float16)y[2];
  p[lane + 96] = (_Float16)y[3];
}

// ---------------------------------------------------------------------------
// int8 quant-dequant on V and store transposed: Vt[(kv*DH + d) * S + s].
// One wave per (s, kv).
// ---------------------------------------------------------------------------
__global__ __launch_bounds__(256) void quant_v_t_kernel(const _Float16* __restrict__ V,
                                                        _Float16* __restrict__ Vt) {
  int wid  = (int)(((size_t)blockIdx.x * blockDim.x + threadIdx.x) >> 5);
  int lane = threadIdx.x & 31;
  int kv = wid & (KVH - 1);
  int s  = wid >> 3;
  const _Float16* p = V + (size_t)s * KVD + kv * DH;

  float y[4];
#pragma unroll
  for (int t = 0; t < 4; ++t) y[t] = (float)p[lane + 32 * t];
  float amax = fmaxf(fmaxf(fabsf(y[0]), fabsf(y[1])), fmaxf(fabsf(y[2]), fabsf(y[3])));
#pragma unroll
  for (int m = 16; m >= 1; m >>= 1) amax = fmaxf(amax, __shfl_xor(amax, m, 32));
  float sc = fmaxf(amax * (1.f / 127.f), 1e-8f);
  float rs = 1.f / sc;
#pragma unroll
  for (int t = 0; t < 4; ++t) {
    float q = fminf(fmaxf(rintf(y[t] * rs), -127.f), 127.f) * sc;
    int d = lane + 32 * t;
    Vt[(size_t)(kv * DH + d) * S_TOK + s] = (_Float16)q;
  }
}

// ---------------------------------------------------------------------------
// Flash attention, causal, GQA (4 Q-heads per KV-head).
// One wave per (head, 16-query tile). Key step = 32.
// Q [S, HD] f16, K [S, KVD] f16, Vt [KVD, S] f16, O [S, HD] f16.
// ---------------------------------------------------------------------------
__global__ __launch_bounds__(32) void attn_kernel(const _Float16* __restrict__ Q,
                                                  const _Float16* __restrict__ Kb,
                                                  const _Float16* __restrict__ Vt,
                                                  _Float16* __restrict__ O) {
  const int q0 = blockIdx.x * 16;
  const int h  = blockIdx.y;
  const int kv = h >> 2;
  const int lane = threadIdx.x;
  const int col = lane & 15;
  const int hi  = lane >> 4;
  const int off = hi * 8;

  __shared__ _Float16 Plds[16 * GLDS];  // P tile relayout, 1.25 KB

  // Resident Q fragments: 16 rows x 128 dims = 4 x (16x32) A-fragments.
  F16x16 qf[4];
  {
    const _Float16* qrow = Q + (size_t)(q0 + col) * HD + h * DH;
#pragma unroll
    for (int c = 0; c < 4; ++c) {
      qf[c].h[0] = *(const h8v*)(qrow + c * 32 + off);
      qf[c].h[1] = *(const h8v*)(qrow + c * 32 + 16 + off);
    }
  }

  float m[8], l[8];
  v8f o_acc[8];
#pragma unroll
  for (int r = 0; r < 8; ++r) { m[r] = -3.0e38f; l[r] = 0.f; }
#pragma unroll
  for (int n = 0; n < 8; ++n) o_acc[n] = v8f_zero();

  const float scale = 0.088388347648318447f;  // 1/sqrt(128)
  const int kend = q0 + 16;

  for (int j0 = 0; j0 < kend; j0 += 32) {
    // ---- scores: two 16x16 tiles (keys j0..j0+15 and j0+16..j0+31) ----
    v8f sacc[2];
    sacc[0] = v8f_zero(); sacc[1] = v8f_zero();
#pragma unroll
    for (int t = 0; t < 2; ++t) {
      int key = j0 + t * 16 + col;
      if (key > S_TOK - 1) key = S_TOK - 1;      // clamped slots are masked below
      const _Float16* krow = Kb + (size_t)key * KVD + kv * DH;
#pragma unroll
      for (int c = 0; c < 4; ++c) {
        F16x16 bf;
        bf.h[0] = *(const h8v*)(krow + c * 32 + off);
        bf.h[1] = *(const h8v*)(krow + c * 32 + 16 + off);
        sacc[t] = wmma_f16(qf[c].v, bf.v, sacc[t]);
      }
    }

    // ---- causal mask + online softmax (per-row, 16-lane reductions) ----
    float p0[8], p1[8], alpha[8];
#pragma unroll
    for (int r = 0; r < 8; ++r) {
      int qi = q0 + r + hi * 8;
      float v0 = sacc[0][r] * scale;
      float v1 = sacc[1][r] * scale;
      if (j0 + col > qi)      v0 = -3.0e38f;
      if (j0 + 16 + col > qi) v1 = -3.0e38f;
      float mx = fmaxf(v0, v1);
#pragma unroll
      for (int d = 8; d >= 1; d >>= 1) mx = fmaxf(mx, __shfl_xor(mx, d, 32));
      float mn = fmaxf(m[r], mx);
      float a  = __expf(m[r] - mn);
      float e0 = __expf(v0 - mn);
      float e1 = __expf(v1 - mn);
      float sum = e0 + e1;
#pragma unroll
      for (int d = 8; d >= 1; d >>= 1) sum += __shfl_xor(sum, d, 32);
      l[r] = l[r] * a + sum;
      m[r] = mn;
      alpha[r] = a;
      p0[r] = e0; p1[r] = e1;
    }
#pragma unroll
    for (int n = 0; n < 8; ++n)
#pragma unroll
      for (int r = 0; r < 8; ++r) o_acc[n][r] *= alpha[r];

    // ---- relayout P (C-layout) -> A-fragment via LDS ----
    __syncthreads();
#pragma unroll
    for (int r = 0; r < 8; ++r) {
      Plds[(r + hi * 8) * GLDS + col]      = (_Float16)p0[r];
      Plds[(r + hi * 8) * GLDS + 16 + col] = (_Float16)p1[r];
    }
    __syncthreads();
    F16x16 pf;
    pf.h[0] = *(const h8v*)&Plds[col * GLDS + off];
    pf.h[1] = *(const h8v*)&Plds[col * GLDS + 16 + off];

    // ---- O += P(16x32) @ V(32 x 128): Vt rows are contiguous in keys ----
    int kb0 = j0 + off;      if (kb0 > S_TOK - 8) kb0 = S_TOK - 8;
    int kb1 = j0 + 16 + off; if (kb1 > S_TOK - 8) kb1 = S_TOK - 8;
#pragma unroll
    for (int n = 0; n < 8; ++n) {
      const _Float16* vrow = Vt + (size_t)(kv * DH + n * 16 + col) * S_TOK;
      F16x16 vf;
      vf.h[0] = *(const h8v*)(vrow + kb0);
      vf.h[1] = *(const h8v*)(vrow + kb1);
      o_acc[n] = wmma_f16(pf.v, vf.v, o_acc[n]);
    }
  }

  // ---- epilogue: normalize and store ----
#pragma unroll
  for (int n = 0; n < 8; ++n) {
#pragma unroll
    for (int r = 0; r < 8; ++r) {
      int row = q0 + r + hi * 8;
      O[(size_t)row * HD + h * DH + n * 16 + col] = (_Float16)(o_acc[n][r] / l[r]);
    }
  }
}

// ---------------------------------------------------------------------------
// Launch
// ---------------------------------------------------------------------------
extern "C" void kernel_launch(void* const* d_in, const int* in_sizes, int n_in,
                              void* d_out, int out_size, void* d_ws, size_t ws_size,
                              hipStream_t stream) {
  const float* hs = (const float*)d_in[0];
  const float* wq = (const float*)d_in[1];
  const float* wk = (const float*)d_in[2];
  const float* wv = (const float*)d_in[3];
  const float* wo = (const float*)d_in[4];
  float* out = (float*)d_out;

  char* ws = (char*)d_ws;
  size_t off = 0;
  _Float16* hs16   = (_Float16*)(ws + off); off += (size_t)S_TOK * D_MODEL * 2;  // 16 MiB
  _Float16* q16    = (_Float16*)(ws + off); off += (size_t)S_TOK * HD * 2;       // 16 MiB
  _Float16* k16    = (_Float16*)(ws + off); off += (size_t)S_TOK * KVD * 2;      //  4 MiB
  _Float16* v16    = (_Float16*)(ws + off); off += (size_t)S_TOK * KVD * 2;      //  4 MiB
  _Float16* vt16   = (_Float16*)(ws + off); off += (size_t)KVD * S_TOK * 2;      //  4 MiB
  _Float16* attn16 = (_Float16*)(ws + off); off += (size_t)S_TOK * HD * 2;       // 16 MiB

  // 0) hidden f32 -> f16
  cvt_f32_f16<<<(S_TOK * D_MODEL / 8) / 256, 256, 0, stream>>>(hs, hs16);

  // 1) QKV projections (f16 WMMA, f32 accumulate)
  gemm_wmma<false><<<dim3(HD / 128, S_TOK / 128), 256, 0, stream>>>(
      hs16, wq, q16, nullptr, D_MODEL, D_MODEL, D_MODEL, HD);
  gemm_wmma<false><<<dim3(KVD / 128, S_TOK / 128), 256, 0, stream>>>(
      hs16, wk, k16, nullptr, D_MODEL, D_MODEL, D_MODEL, KVD);
  gemm_wmma<false><<<dim3(KVD / 128, S_TOK / 128), 256, 0, stream>>>(
      hs16, wv, v16, nullptr, D_MODEL, D_MODEL, D_MODEL, KVD);

  // 2) RoPE(Q); RoPE + int8 qdq (K); int8 qdq + transpose (V)
  rope_q_kernel<<<(S_TOK * HQ * 64) / 256, 256, 0, stream>>>(q16);
  rope_quant_k_kernel<<<(S_TOK * KVH * 32) / 256, 256, 0, stream>>>(k16);
  quant_v_t_kernel<<<(S_TOK * KVH * 32) / 256, 256, 0, stream>>>(v16, vt16);

  // 3) causal flash attention (GQA)
  attn_kernel<<<dim3(S_TOK / 16, HQ), 32, 0, stream>>>(q16, k16, vt16, attn16);

  // 4) output projection -> f32 d_out
  gemm_wmma<true><<<dim3(D_MODEL / 128, S_TOK / 128), 256, 0, stream>>>(
      attn16, wo, nullptr, out, HD, HD, HD, D_MODEL);
}
